// MultiAttention_44839458570473
// MI455X (gfx1250) — compile-verified, matrix operand order checked by reading, not run on hardware
//
#include <hip/hip_runtime.h>
#include <hip/hip_bf16.h>
#include <math.h>

typedef __attribute__((ext_vector_type(16))) _Float16 v16h;
typedef __attribute__((ext_vector_type(8)))  _Float16 v8h;
typedef __attribute__((ext_vector_type(8)))  float    v8f;

#define N_TOK 4096
#define C_IN  64
#define NHEAD 8
#define DKH   64
#define DMODEL 512

// ---- WMMA A/B element -> K mapping (16-bit A 16x32 layout, CDNA5 ISA 7.12.2) ----
// element e (0..15) of lane; g = lane>>4 :  K = (e&7) + 8*(2*(e>>3) + g)
// => chunks e=0..7 -> K = 8g..8g+7 (contiguous), e=8..15 -> K = 16+8g..23+8g
__device__ __forceinline__ int kmap(int e, int g) {
    return (e & 7) + (((e >> 3) * 2 + g) * 8);
}

// combine two contiguous 8-half (16B) chunks into a v16h WMMA operand
__device__ __forceinline__ v16h ld_op(const _Float16* base, int g) {
    v8h lo = *(const v8h*)(base + 8 * g);
    v8h hi = *(const v8h*)(base + 16 + 8 * g);
    v16h r;
#pragma unroll
    for (int e = 0; e < 8; ++e) { r[e] = lo[e]; r[e + 8] = hi[e]; }
    return r;
}

__device__ __forceinline__ float gelu_f(float x) {
    return 0.5f * x * (1.0f + erff(x * 0.70710678118654752f));
}
__device__ __forceinline__ float sigmoid_f(float x) {
    return 1.0f / (1.0f + __expf(-x));
}

// ---------------------------------------------------------------------------
// 1a) transpose + convert x_in [C, n] -> x_h [n, C] f16
__global__ void k_prep_x(const float* __restrict__ x_in, _Float16* __restrict__ x_h) {
    int idx = blockIdx.x * blockDim.x + threadIdx.x;   // 262144
    int c = idx & 63, p = idx >> 6;
    x_h[idx] = (_Float16)x_in[c * N_TOK + p];
}

// 1b) convert GEMM weights to f16 (one shot)
__global__ void k_prep_w(const float* __restrict__ Wq, const float* __restrict__ Wk,
                         const float* __restrict__ Wv, const float* __restrict__ Wout,
                         _Float16* __restrict__ wq_h, _Float16* __restrict__ wk_h,
                         _Float16* __restrict__ wv_h, _Float16* __restrict__ wout_h) {
    int idx = blockIdx.x * blockDim.x + threadIdx.x;   // 131072
    if (idx < 32768)        wq_h[idx] = (_Float16)Wq[idx];
    else if (idx < 65536)   wk_h[idx - 32768] = (_Float16)Wk[idx - 32768];
    else if (idx < 98304)   wv_h[idx - 65536] = (_Float16)Wv[idx - 65536];
    else                    wout_h[idx - 98304] = (_Float16)Wout[idx - 98304];
}

// ---------------------------------------------------------------------------
// 2) QKV GEMM:  out[p,d] = sum_c x[p,c] * W[d,c]   (M=4096, N=512, K=64)
//    one wave -> one 16x16 tile via 2 chained v_wmma_f32_16x16x32_f16
__global__ void k_qkv_gemm(const _Float16* __restrict__ x_h,
                           const _Float16* __restrict__ wq_h,
                           const _Float16* __restrict__ wk_h,
                           const _Float16* __restrict__ wv_h,
                           float* __restrict__ qraw,
                           float* __restrict__ kraw,
                           float* __restrict__ vraw) {
    const _Float16* W = (blockIdx.z == 0) ? wq_h : (blockIdx.z == 1) ? wk_h : wv_h;
    float* O          = (blockIdx.z == 0) ? qraw : (blockIdx.z == 1) ? kraw : vraw;
    int i0 = blockIdx.x * 16;           // token tile
    int n0 = blockIdx.y * 16;           // d tile
    int lane = threadIdx.x, m = lane & 15, g = lane >> 4;

    v8f acc = {};
#pragma unroll
    for (int kk = 0; kk < 2; ++kk) {
        v16h a = ld_op(x_h + (i0 + m) * 64 + kk * 32, g);
        v16h b = ld_op(W   + (n0 + m) * 64 + kk * 32, g);   // lane's N index == m
        acc = __builtin_amdgcn_wmma_f32_16x16x32_f16(false, a, false, b,
                                                     (short)0, acc, false, false);
    }
#pragma unroll
    for (int r = 0; r < 8; ++r)
        O[(i0 + g * 8 + r) * DMODEL + n0 + m] = acc[r];
}

// ---------------------------------------------------------------------------
// 3) cosine-normalize q,k per (head, token); write f16 [h][p][64]
__global__ void k_norm_qk(const float* __restrict__ qraw, const float* __restrict__ kraw,
                          _Float16* __restrict__ qh, _Float16* __restrict__ kh) {
    int b = blockIdx.x;               // h*4096 + p
    int h = b >> 12, p = b & 4095;
    int lane = threadIdx.x;           // 32
    int base = p * DMODEL + h * DKH;
    float q0 = qraw[base + lane], q1 = qraw[base + 32 + lane];
    float k0 = kraw[base + lane], k1 = kraw[base + 32 + lane];
    float sq = q0 * q0 + q1 * q1;
    float sk = k0 * k0 + k1 * k1;
#pragma unroll
    for (int off = 16; off > 0; off >>= 1) {
        sq += __shfl_xor(sq, off, 32);
        sk += __shfl_xor(sk, off, 32);
    }
    float dq = fmaxf(sqrtf(sq), 1e-12f);
    float dk = fmaxf(sqrtf(sk), 1e-12f);
    int ob = (h * N_TOK + p) * DKH;
    qh[ob + lane]      = (_Float16)(q0 / dq);
    qh[ob + 32 + lane] = (_Float16)(q1 / dq);
    kh[ob + lane]      = (_Float16)(k0 / dk);
    kh[ob + 32 + lane] = (_Float16)(k1 / dk);
}

// ---------------------------------------------------------------------------
// 4) v split: f16 v_t [d][p] (d-major, for PV B-matrix b128 loads) and
//    f32 v_img [d][p] for conv branch
__global__ void k_vsplit(const float* __restrict__ vraw,
                         _Float16* __restrict__ v_t, float* __restrict__ v_img) {
    int idx = blockIdx.x * blockDim.x + threadIdx.x;  // 2M : p*512+d
    int p = idx >> 9, d = idx & 511;
    float v = vraw[idx];
    v_t[d * N_TOK + p]   = (_Float16)v;
    v_img[d * N_TOK + p] = v;
}

// ---------------------------------------------------------------------------
// 5) flash attention: one wave per (16-row tile, head). Online softmax.
__global__ void k_flash(const _Float16* __restrict__ qh, const _Float16* __restrict__ kh,
                        const _Float16* __restrict__ v_t, const float* __restrict__ scale,
                        float* __restrict__ att) {
    int hId = blockIdx.y;
    int i0  = blockIdx.x * 16;
    int lane = threadIdx.x, m = lane & 15, g = lane >> 4;
    float sc = scale[hId];
    __shared__ _Float16 plds[16 * 32];

    const _Float16* qbase = qh + (hId * N_TOK + i0) * DKH;
    const _Float16* kbase = kh + hId * N_TOK * DKH;
    const _Float16* vbase = v_t + hId * DKH * N_TOK;   // [d_local][p]

    v16h aq[2];
#pragma unroll
    for (int kk = 0; kk < 2; ++kk)
        aq[kk] = ld_op(qbase + m * DKH + kk * 32, g);

    v8f acc0 = {}, acc1 = {}, acc2 = {}, acc3 = {};
    float mrow[8], lrow[8];
#pragma unroll
    for (int r = 0; r < 8; ++r) { mrow[r] = -3.0e38f; lrow[r] = 0.f; }

    for (int jb = 0; jb < N_TOK; jb += 32) {
        v8f s0 = {}, s1 = {};
#pragma unroll
        for (int sub = 0; sub < 2; ++sub) {
            int j0 = jb + sub * 16;
            v8f st = {};
#pragma unroll
            for (int kk = 0; kk < 2; ++kk) {
                v16h b = ld_op(kbase + (j0 + m) * DKH + kk * 32, g);
                st = __builtin_amdgcn_wmma_f32_16x16x32_f16(false, aq[kk], false, b,
                                                            (short)0, st, false, false);
            }
            if (sub == 0) s0 = st; else s1 = st;
        }
        // online softmax across the 32-column block
#pragma unroll
        for (int r = 0; r < 8; ++r) {
            float a0 = s0[r] * sc, a1 = s1[r] * sc;
            float tm = fmaxf(a0, a1);
#pragma unroll
            for (int off = 8; off > 0; off >>= 1) tm = fmaxf(tm, __shfl_xor(tm, off, 32));
            float mn = fmaxf(mrow[r], tm);
            float alpha = __expf(mrow[r] - mn);
            float p0 = __expf(a0 - mn), p1 = __expf(a1 - mn);
            float rs = p0 + p1;
#pragma unroll
            for (int off = 8; off > 0; off >>= 1) rs += __shfl_xor(rs, off, 32);
            lrow[r] = lrow[r] * alpha + rs;
            mrow[r] = mn;
            acc0[r] *= alpha; acc1[r] *= alpha; acc2[r] *= alpha; acc3[r] *= alpha;
            plds[(g * 8 + r) * 32 + m]      = (_Float16)p0;
            plds[(g * 8 + r) * 32 + 16 + m] = (_Float16)p1;
        }
        __syncthreads();
        v16h ap;                               // P tile in A-layout (K = local j)
#pragma unroll
        for (int e = 0; e < 16; ++e) ap[e] = plds[m * 32 + kmap(e, g)];
        __syncthreads();
#pragma unroll
        for (int t = 0; t < 4; ++t) {
            // B[j][d]: lane -> d = t*16+m; contiguous j chunks -> b128 loads
            v16h b = ld_op(vbase + (t * 16 + m) * N_TOK + jb, g);
            v8f* ap8 = (t == 0) ? &acc0 : (t == 1) ? &acc1 : (t == 2) ? &acc2 : &acc3;
            *ap8 = __builtin_amdgcn_wmma_f32_16x16x32_f16(false, ap, false, b,
                                                          (short)0, *ap8, false, false);
        }
    }
#pragma unroll
    for (int r = 0; r < 8; ++r) {
        float inv = 1.f / lrow[r];
        int prow = i0 + g * 8 + r;
        int ob = prow * DMODEL + hId * DKH;
        att[ob +  0 + m] = acc0[r] * inv;
        att[ob + 16 + m] = acc1[r] * inv;
        att[ob + 32 + m] = acc2[r] * inv;
        att[ob + 48 + m] = acc3[r] * inv;
    }
}

// ---------------------------------------------------------------------------
// 6) depthwise 3x3 conv + bias + BN + GELU on v_img -> conv_x [d][p]
__global__ void k_conv_branch(const float* __restrict__ v_img,
                              const float* __restrict__ dw_w, const float* __restrict__ dw_b,
                              const float* __restrict__ dw_g, const float* __restrict__ dw_bt,
                              const float* __restrict__ dw_m, const float* __restrict__ dw_v,
                              float* __restrict__ conv_x) {
    int idx = blockIdx.x * blockDim.x + threadIdx.x;  // 2M : d*4096+p
    int d = idx >> 12, p = idx & 4095;
    int hh = p >> 6, ww = p & 63;
    float acc = 0.f;
#pragma unroll
    for (int ky = 0; ky < 3; ++ky)
#pragma unroll
        for (int kx = 0; kx < 3; ++kx) {
            int yy = hh + ky - 1, xx = ww + kx - 1;
            if ((unsigned)yy < 64u && (unsigned)xx < 64u)
                acc += v_img[d * N_TOK + yy * 64 + xx] * dw_w[d * 9 + ky * 3 + kx];
        }
    acc += dw_b[d];
    float s = dw_g[d] * rsqrtf(dw_v[d] + 1e-5f);
    acc = (acc - dw_m[d]) * s + dw_bt[d];
    conv_x[idx] = gelu_f(acc);
}

// ---------------------------------------------------------------------------
// 7) pooled[d] = mean_p att[p][d]
__global__ void k_pool(const float* __restrict__ att, float* __restrict__ pooled) {
    int d = blockIdx.x, t = threadIdx.x;       // 512 blocks x 256
    float s = 0.f;
    for (int p = t; p < N_TOK; p += 256) s += att[p * DMODEL + d];
    __shared__ float red[256];
    red[t] = s; __syncthreads();
    for (int o = 128; o > 0; o >>= 1) { if (t < o) red[t] += red[t + o]; __syncthreads(); }
    if (t == 0) pooled[d] = red[0] * (1.f / (float)N_TOK);
}

// ---------------------------------------------------------------------------
// 8) channel interaction -> sigmoid gate per channel
__global__ void k_ci(const float* __restrict__ pooled,
                     const float* __restrict__ w1, const float* __restrict__ b1,
                     const float* __restrict__ w2, const float* __restrict__ b2,
                     float* __restrict__ gate_c) {
    __shared__ float h1[128];
    int t = threadIdx.x;                       // 512
    if (t < 128) {
        float a = b1[t];
        for (int d = 0; d < DMODEL; ++d) a += pooled[d] * w1[t * DMODEL + d];
        h1[t] = gelu_f(a);
    }
    __syncthreads();
    float a = b2[t];
    for (int e = 0; e < 128; ++e) a += h1[e] * w2[t * 128 + e];
    gate_c[t] = sigmoid_f(a);
}

// ---------------------------------------------------------------------------
// 9) spatial interaction -> sigmoid gate per pixel
__global__ void k_spatial(const float* __restrict__ conv_x,
                          const float* __restrict__ w1, const float* __restrict__ b1,
                          const float* __restrict__ gg, const float* __restrict__ bt,
                          const float* __restrict__ mm, const float* __restrict__ vv,
                          const float* __restrict__ w2, const float* __restrict__ b2,
                          float* __restrict__ gate_s) {
    int p = blockIdx.x * blockDim.x + threadIdx.x;   // 4096
    float sm = b2[0];
    for (int e = 0; e < 32; ++e) {
        float a = b1[e];
        for (int d = 0; d < DMODEL; ++d) a += conv_x[d * N_TOK + p] * w1[e * DMODEL + d];
        float s = gg[e] * rsqrtf(vv[e] + 1e-5f);
        a = (a - mm[e]) * s + bt[e];
        sm += gelu_f(a) * w2[e];
    }
    gate_s[p] = sigmoid_f(sm);
}

// ---------------------------------------------------------------------------
// 10) z = att * gate_s[p] + conv_x^T * gate_c[d]  -> f16 [p][512]
__global__ void k_combine(const float* __restrict__ att, const float* __restrict__ conv_x,
                          const float* __restrict__ gate_s, const float* __restrict__ gate_c,
                          _Float16* __restrict__ zh) {
    int idx = blockIdx.x * blockDim.x + threadIdx.x;  // 2M : p*512+d
    int p = idx >> 9, d = idx & 511;
    float z = att[idx] * gate_s[p] + conv_x[d * N_TOK + p] * gate_c[d];
    zh[idx] = (_Float16)z;
}

// ---------------------------------------------------------------------------
// 11) output projection: y[p,c] = sum_d z[p,d]*Wout[c,d] + bout[c]   (WMMA)
__global__ void k_out_gemm(const _Float16* __restrict__ zh, const _Float16* __restrict__ wout_h,
                           const float* __restrict__ bout, float* __restrict__ y) {
    int i0 = blockIdx.x * 16, n0 = blockIdx.y * 16;
    int lane = threadIdx.x, m = lane & 15, g = lane >> 4;
    v8f acc = {};
#pragma unroll
    for (int kk = 0; kk < 16; ++kk) {
        v16h a = ld_op(zh     + (i0 + m) * DMODEL + kk * 32, g);
        v16h b = ld_op(wout_h + (n0 + m) * DMODEL + kk * 32, g);
        acc = __builtin_amdgcn_wmma_f32_16x16x32_f16(false, a, false, b,
                                                     (short)0, acc, false, false);
    }
    float bo = bout[n0 + m];
#pragma unroll
    for (int r = 0; r < 8; ++r)
        y[(i0 + g * 8 + r) * C_IN + n0 + m] = acc[r] + bo;
}

// ---------------------------------------------------------------------------
// 12) posembed stage 1: t = gelu(dwconv3x3(x_in, pe_w1))
__global__ void k_embed1(const float* __restrict__ x_in, const float* __restrict__ pe_w1,
                         float* __restrict__ t_emb) {
    int idx = blockIdx.x * blockDim.x + threadIdx.x;  // 262144 : c*4096+p
    int c = idx >> 12, p = idx & 4095;
    int hh = p >> 6, ww = p & 63;
    float acc = 0.f;
#pragma unroll
    for (int ky = 0; ky < 3; ++ky)
#pragma unroll
        for (int kx = 0; kx < 3; ++kx) {
            int yy = hh + ky - 1, xx = ww + kx - 1;
            if ((unsigned)yy < 64u && (unsigned)xx < 64u)
                acc += x_in[c * N_TOK + yy * 64 + xx] * pe_w1[c * 9 + ky * 3 + kx];
        }
    t_emb[idx] = gelu_f(acc);
}

// 13) posembed stage 2 + final add (out_flat index == y_flat index)
__global__ void k_embed2(const float* __restrict__ t_emb, const float* __restrict__ pe_w2,
                         const float* __restrict__ y, float* __restrict__ out) {
    int idx = blockIdx.x * blockDim.x + threadIdx.x;  // 262144
    int c = idx >> 12, p = idx & 4095;
    int hh = p >> 6, ww = p & 63;
    float acc = 0.f;
#pragma unroll
    for (int ky = 0; ky < 3; ++ky)
#pragma unroll
        for (int kx = 0; kx < 3; ++kx) {
            int yy = hh + ky - 1, xx = ww + kx - 1;
            if ((unsigned)yy < 64u && (unsigned)xx < 64u)
                acc += t_emb[c * N_TOK + yy * 64 + xx] * pe_w2[c * 9 + ky * 3 + kx];
        }
    out[idx] = y[idx] + acc;
}

// ---------------------------------------------------------------------------
extern "C" void kernel_launch(void* const* d_in, const int* in_sizes, int n_in,
                              void* d_out, int out_size, void* d_ws, size_t ws_size,
                              hipStream_t stream) {
    (void)in_sizes; (void)n_in; (void)out_size;
    const float* x_in  = (const float*)d_in[0];
    const float* Wq    = (const float*)d_in[1];
    const float* Wk    = (const float*)d_in[2];
    const float* Wv    = (const float*)d_in[3];
    const float* scale = (const float*)d_in[4];
    const float* Wout  = (const float*)d_in[5];
    const float* bout  = (const float*)d_in[6];
    const float* dw_w  = (const float*)d_in[7];
    const float* dw_b  = (const float*)d_in[8];
    const float* dw_g  = (const float*)d_in[9];
    const float* dw_bt = (const float*)d_in[10];
    const float* dw_m  = (const float*)d_in[11];
    const float* dw_v  = (const float*)d_in[12];
    const float* ci_w1 = (const float*)d_in[13];
    const float* ci_b1 = (const float*)d_in[14];
    const float* ci_w2 = (const float*)d_in[15];
    const float* ci_b2 = (const float*)d_in[16];
    const float* si_w1 = (const float*)d_in[17];
    const float* si_b1 = (const float*)d_in[18];
    const float* si_g  = (const float*)d_in[19];
    const float* si_bt = (const float*)d_in[20];
    const float* si_m  = (const float*)d_in[21];
    const float* si_v  = (const float*)d_in[22];
    const float* si_w2 = (const float*)d_in[23];
    const float* si_b2 = (const float*)d_in[24];
    const float* pe_w1 = (const float*)d_in[25];
    const float* pe_w2 = (const float*)d_in[26];
    float* out = (float*)d_out;

    // workspace layout (all offsets 256B aligned)
    char* ws = (char*)d_ws;
    size_t off = 0;
    _Float16* x_h    = (_Float16*)(ws + off); off += (size_t)N_TOK * C_IN * 2;        // 512 KB
    _Float16* wq_h   = (_Float16*)(ws + off); off += (size_t)DMODEL * C_IN * 2;       // 64 KB
    _Float16* wk_h   = (_Float16*)(ws + off); off += (size_t)DMODEL * C_IN * 2;
    _Float16* wv_h   = (_Float16*)(ws + off); off += (size_t)DMODEL * C_IN * 2;
    _Float16* wout_h = (_Float16*)(ws + off); off += (size_t)C_IN * DMODEL * 2;
    float*    qraw   = (float*)(ws + off);    off += (size_t)N_TOK * DMODEL * 4;      // 8 MB
    float*    kraw   = (float*)(ws + off);    off += (size_t)N_TOK * DMODEL * 4;
    float*    vraw   = (float*)(ws + off);    off += (size_t)N_TOK * DMODEL * 4;
    _Float16* qh     = (_Float16*)(ws + off); off += (size_t)N_TOK * DMODEL * 2;      // 4 MB
    _Float16* kh     = (_Float16*)(ws + off); off += (size_t)N_TOK * DMODEL * 2;
    _Float16* v_t    = (_Float16*)(ws + off); off += (size_t)N_TOK * DMODEL * 2;
    float*    v_img  = (float*)(ws + off);    off += (size_t)N_TOK * DMODEL * 4;
    float*    att    = (float*)(ws + off);    off += (size_t)N_TOK * DMODEL * 4;
    float*    conv_x = (float*)(ws + off);    off += (size_t)N_TOK * DMODEL * 4;
    float*    pooled = (float*)(ws + off);    off += 2048;
    float*    gate_c = (float*)(ws + off);    off += 2048;
    float*    gate_s = (float*)(ws + off);    off += (size_t)N_TOK * 4;               // 16 KB
    _Float16* zh     = (_Float16*)(ws + off); off += (size_t)N_TOK * DMODEL * 2;
    float*    ybuf   = (float*)(ws + off);    off += (size_t)N_TOK * C_IN * 4;
    float*    t_emb  = (float*)(ws + off);    off += (size_t)N_TOK * C_IN * 4;
    if (ws_size < off) return;

    k_prep_x     <<<1024, 256, 0, stream>>>(x_in, x_h);
    k_prep_w     <<<512, 256, 0, stream>>>(Wq, Wk, Wv, Wout, wq_h, wk_h, wv_h, wout_h);
    k_qkv_gemm   <<<dim3(256, 32, 3), 32, 0, stream>>>(x_h, wq_h, wk_h, wv_h, qraw, kraw, vraw);
    k_norm_qk    <<<NHEAD * N_TOK, 32, 0, stream>>>(qraw, kraw, qh, kh);
    k_vsplit     <<<8192, 256, 0, stream>>>(vraw, v_t, v_img);
    k_flash      <<<dim3(256, NHEAD), 32, 0, stream>>>(qh, kh, v_t, scale, att);
    k_conv_branch<<<8192, 256, 0, stream>>>(v_img, dw_w, dw_b, dw_g, dw_bt, dw_m, dw_v, conv_x);
    k_pool       <<<DMODEL, 256, 0, stream>>>(att, pooled);
    k_ci         <<<1, 512, 0, stream>>>(pooled, ci_w1, ci_b1, ci_w2, ci_b2, gate_c);
    k_spatial    <<<16, 256, 0, stream>>>(conv_x, si_w1, si_b1, si_g, si_bt, si_m, si_v,
                                          si_w2, si_b2, gate_s);
    k_combine    <<<8192, 256, 0, stream>>>(att, conv_x, gate_s, gate_c, zh);
    k_out_gemm   <<<dim3(256, 4), 32, 0, stream>>>(zh, wout_h, bout, ybuf);
    k_embed1     <<<1024, 256, 0, stream>>>(x_in, pe_w1, t_emb);
    k_embed2     <<<1024, 256, 0, stream>>>(t_emb, pe_w2, ybuf, out);
}
